// AttentionBlock_27376121545458
// MI455X (gfx1250) — matrix operand order repaired and checked by reading, not compiled, not run on hardware
//
#include <hip/hip_runtime.h>
#include <hip/hip_bf16.h>

typedef _Float16 half_t;
typedef __attribute__((ext_vector_type(16))) _Float16 v16h;
typedef __attribute__((ext_vector_type(8)))  _Float16 v8h;
typedef __attribute__((ext_vector_type(8)))  float    v8f;
typedef __attribute__((ext_vector_type(4)))  unsigned u32x4;
typedef __attribute__((ext_vector_type(8)))  unsigned u32x8;

#define BB 32
#define CC 512
#define NN 1024            // H*W
#define OO 1536            // 3*C
#define GG 32
#define CG 16              // C / GROUPS
#define EPSV 1e-5f
#define QK_SCALE 0.21022410381342863f   // 512^(-1/4)

struct F16Pair { v8h lo, hi; };

static __device__ inline v16h pack_frag(v8h lo, v8h hi) {
    F16Pair f; f.lo = lo; f.hi = hi;
    return __builtin_bit_cast(v16h, f);
}

// ---- A fragment (16x32 f16, M x K), A row-major, tile base = &A[m0*lda + k0]
static __device__ inline v16h load_a_f16(const half_t* Abase, int lda) {
    int lane = threadIdx.x & 31;
    int row  = lane & 15;
    int kh   = (lane >> 4) * 8;
    const half_t* p = Abase + (size_t)row * lda + kh;
    v8h lo = *(const v8h*)(p);
    v8h hi = *(const v8h*)(p + 16);
    return pack_frag(lo, hi);
}

// ---- B fragment (32x16, K x N) where logical B[kk][n] = Bt[n*ldb + kk]
static __device__ inline v16h load_b_t(const half_t* Btbase, int ldb) {
    int lane = threadIdx.x & 31;
    int n  = lane & 15;
    int kb = (lane >> 4) * 8;
    const half_t* p = Btbase + (size_t)n * ldb + kb;
    v8h lo = *(const v8h*)(p);
    v8h hi = *(const v8h*)(p + 16);
    return pack_frag(lo, hi);
}

static __device__ inline v8f wmma_f16(v16h a, v16h b, v8f c) {
    return __builtin_amdgcn_wmma_f32_16x16x32_f16(false, a, false, b, (short)0, c, false, false);
}

// ---- Async global->LDS: issue a 64-row x 256-half panel copy (32 KB), 256 threads.
// Thread tid copies 128 contiguous bytes; LDS layout is row-major [64][256] halfs.
static __device__ inline void async_issue_panel(half_t* sB, const half_t* gsrc, int ld_global) {
    int tid = threadIdx.x;
    int row = tid >> 2;
    int col = (tid & 3) * 64;
    const half_t* g = gsrc + (size_t)row * ld_global + col;
    unsigned lds = (unsigned)(uintptr_t)sB + (unsigned)(tid * 128);
#pragma unroll
    for (int i = 0; i < 8; ++i) {
        asm volatile("global_load_async_to_lds_b128 %0, %1, off"
                     :: "v"(lds + i * 16), "v"(g + i * 8) : "memory");
    }
}
static __device__ inline void async_wait() {
    asm volatile("s_wait_asynccnt 0x0" ::: "memory");
}

// ---- Tensor Data Mover: issue a 2D tile DMA (64 rows x 256 halfs) from one wave.
// D# packing per CDNA5 ISA ch.8.
static __device__ inline void tdm_issue_panel(half_t* sB, const half_t* gsrc, unsigned ld_global) {
    unsigned long long ga = (unsigned long long)(uintptr_t)gsrc;
    u32x4 g0;
    g0[0] = 1u;                                            // count=1, user mode
    g0[1] = (unsigned)(uintptr_t)sB;                       // lds_addr (bytes)
    g0[2] = (unsigned)ga;                                  // global_addr[31:0]
    g0[3] = (unsigned)((ga >> 32) & 0x01ffffffu) | (2u << 30); // addr[56:32] | type=2
    u32x8 g1;
    g1[0] = 1u << 16;                                      // wg_mask=0, data_size=1 (2 bytes)
    g1[1] = (ld_global & 0xffffu) << 16;                   // tensor_dim0 lo16
    g1[2] = (ld_global >> 16) | (64u << 16);               // tensor_dim0 hi16 | tensor_dim1 lo16
    g1[3] = (256u << 16);                                  // tensor_dim1 hi16=0 | tile_dim0=256
    g1[4] = 64u;                                           // tile_dim1=64, tile_dim2=0
    g1[5] = ld_global;                                     // tensor_dim0_stride lo32
    g1[6] = 0u;
    g1[7] = 0u;
    asm volatile("tensor_load_to_lds %0, %1" :: "s"(g0), "s"(g1) : "memory");
}
static __device__ inline void tdm_wait() {
    asm volatile("s_wait_tensorcnt 0x0" ::: "memory");
}

// =================== Kernel 0: weight conversion to f16 ===================
__global__ void k_cvt_weights(const float* __restrict__ qkv_w, const float* __restrict__ proj_w,
                              half_t* __restrict__ wq, half_t* __restrict__ wp) {
    int i = blockIdx.x * blockDim.x + threadIdx.x;
    if (i < OO * CC) wq[i] = (half_t)qkv_w[i];
    if (i < CC * CC) wp[i] = (half_t)proj_w[i];
}

// =================== Kernel 1: GroupNorm -> transposed [B,N,C] f16 ===================
__global__ void __launch_bounds__(256) k_groupnorm(const float* __restrict__ x,
                                                   const float* __restrict__ gamma,
                                                   const float* __restrict__ beta,
                                                   half_t* __restrict__ hn) {
    int b = blockIdx.x / GG;
    int g = blockIdx.x % GG;
    const float* xb = x + (size_t)b * CC * NN + (size_t)g * CG * NN;
    __shared__ float s1[256], s2[256];
    int tid = threadIdx.x;
    float sum = 0.f, sq = 0.f;
    for (int i = tid; i < CG * NN; i += 256) {
        float v = xb[i];
        sum += v; sq += v * v;
    }
    s1[tid] = sum; s2[tid] = sq;
    __syncthreads();
    for (int off = 128; off > 0; off >>= 1) {
        if (tid < off) { s1[tid] += s1[tid + off]; s2[tid] += s2[tid + off]; }
        __syncthreads();
    }
    const float invN = 1.0f / (float)(CG * NN);
    float mean = s1[0] * invN;
    float var  = s2[0] * invN - mean * mean;
    float inv  = rsqrtf(var + EPSV);
    for (int i = tid; i < CG * NN; i += 256) {
        int cl = i >> 10;
        int p  = i & (NN - 1);
        int c  = g * CG + cl;
        float v = (xb[i] - mean) * inv * gamma[c] + beta[c];
        hn[(size_t)b * NN * CC + (size_t)p * CC + c] = (half_t)v;
    }
}

// =================== Kernel 2: QKV GEMM (double-buffered TDM weight panels) ===================
__global__ void __launch_bounds__(256) k_qkv(const half_t* __restrict__ hn,
                                             const half_t* __restrict__ wq,
                                             const float* __restrict__ qkv_b,
                                             half_t* __restrict__ qb,
                                             half_t* __restrict__ kb,
                                             half_t* __restrict__ vb) {
    __shared__ half_t sB[2][64 * 256];              // 2 x 32 KB double-buffered panel
    const int NT = OO / 64;                         // 24
    int nt  = blockIdx.x % NT;
    int mtg = (blockIdx.x / NT) & 7;
    int b   = blockIdx.x / (NT * 8);
    int wv  = threadIdx.x >> 5;
    int m0  = (mtg * 8 + wv) * 16;
    int n0  = nt * 64;

    const half_t* A    = hn + (size_t)b * NN * CC + (size_t)m0 * CC;
    const half_t* wsrc = wq + (size_t)n0 * CC;

    if (wv == 0) tdm_issue_panel(sB[0], wsrc, CC);
    v8f acc[4] = {};
    for (int h = 0; h < 2; ++h) {
        if (wv == 0) tdm_wait();
        __syncthreads();
        if (h + 1 < 2 && wv == 0) tdm_issue_panel(sB[(h + 1) & 1], wsrc + (h + 1) * 256, CC);
        const half_t* Ah = A + h * 256;
        const half_t* Bp = sB[h & 1];
#pragma unroll 4
        for (int kk = 0; kk < 256; kk += 32) {
            v16h a = load_a_f16(Ah + kk, CC);
#pragma unroll
            for (int t = 0; t < 4; ++t) {
                v16h bf = load_b_t(Bp + (size_t)(t * 16) * 256 + kk, 256);
                acc[t] = wmma_f16(a, bf, acc[t]);
            }
        }
    }
    int lane = threadIdx.x & 31;
    int n = lane & 15, mb = (lane >> 4) * 8;
#pragma unroll
    for (int t = 0; t < 4; ++t) {
        int o = n0 + t * 16 + n;
        float bias = qkv_b[o];
#pragma unroll
        for (int rr = 0; rr < 8; ++rr) {
            int p = m0 + mb + rr;
            float val = acc[t][rr] + bias;
            if (o < CC)
                qb[(size_t)b * NN * CC + (size_t)p * CC + o] = (half_t)(val * QK_SCALE);
            else if (o < 2 * CC)
                kb[(size_t)b * NN * CC + (size_t)p * CC + (o - CC)] = (half_t)(val * QK_SCALE);
            else
                vb[(size_t)b * CC * NN + (size_t)(o - 2 * CC) * NN + p] = (half_t)val;  // V^T
        }
    }
}

// =================== Kernel 3: S = Q.K^T (double-buffered async K panels) ===================
__global__ void __launch_bounds__(256) k_scores(const half_t* __restrict__ qb,
                                                const half_t* __restrict__ kb,
                                                float* __restrict__ S) {
    __shared__ half_t sB[2][64 * 256];
    const int NT = NN / 64;                         // 16
    int nt  = blockIdx.x % NT;
    int mtg = (blockIdx.x / NT) & 7;
    int b   = blockIdx.x / (NT * 8);
    int wv  = threadIdx.x >> 5;
    int m0  = (mtg * 8 + wv) * 16;
    int n0  = nt * 64;

    const half_t* A   = qb + (size_t)b * NN * CC + (size_t)m0 * CC;
    const half_t* src = kb + (size_t)b * NN * CC + (size_t)n0 * CC;

    async_issue_panel(sB[0], src, CC);
    v8f acc[4] = {};
    for (int h = 0; h < 2; ++h) {
        async_wait();
        __syncthreads();
        if (h + 1 < 2) async_issue_panel(sB[(h + 1) & 1], src + (h + 1) * 256, CC);
        const half_t* Ah = A + h * 256;
        const half_t* Bp = sB[h & 1];
#pragma unroll 4
        for (int kk = 0; kk < 256; kk += 32) {
            v16h a = load_a_f16(Ah + kk, CC);
#pragma unroll
            for (int t = 0; t < 4; ++t) {
                v16h bf = load_b_t(Bp + (size_t)(t * 16) * 256 + kk, 256);
                acc[t] = wmma_f16(a, bf, acc[t]);
            }
        }
    }
    int lane = threadIdx.x & 31;
    int n = lane & 15, mb = (lane >> 4) * 8;
    float* Sb = S + (size_t)b * NN * NN;
#pragma unroll
    for (int t = 0; t < 4; ++t) {
#pragma unroll
        for (int rr = 0; rr < 8; ++rr) {
            Sb[(size_t)(m0 + mb + rr) * NN + (n0 + t * 16 + n)] = acc[t][rr];
        }
    }
}

// =================== Kernel 4: row softmax -> f16 probabilities ===================
__global__ void __launch_bounds__(256) k_softmax(const float* __restrict__ S,
                                                 half_t* __restrict__ P) {
    const float* p = S + (size_t)blockIdx.x * NN;
    half_t* po = P + (size_t)blockIdx.x * NN;
    __shared__ float sm[256];
    int tid = threadIdx.x;
    float v[4];
    float m = -3.402823466e38f;
#pragma unroll
    for (int i = 0; i < 4; ++i) { v[i] = p[tid + i * 256]; m = fmaxf(m, v[i]); }
    sm[tid] = m;
    __syncthreads();
    for (int off = 128; off > 0; off >>= 1) {
        if (tid < off) sm[tid] = fmaxf(sm[tid], sm[tid + off]);
        __syncthreads();
    }
    float rowmax = sm[0];
    __syncthreads();
    float s = 0.f;
#pragma unroll
    for (int i = 0; i < 4; ++i) { v[i] = __expf(v[i] - rowmax); s += v[i]; }
    sm[tid] = s;
    __syncthreads();
    for (int off = 128; off > 0; off >>= 1) {
        if (tid < off) sm[tid] += sm[tid + off];
        __syncthreads();
    }
    float inv = 1.0f / sm[0];
#pragma unroll
    for (int i = 0; i < 4; ++i) po[tid + i * 256] = (half_t)(v[i] * inv);
}

// =================== Kernel 5: H = P.V (double-buffered async V^T panels) ===================
__global__ void __launch_bounds__(256) k_pv(const half_t* __restrict__ P,
                                            const half_t* __restrict__ vb,   // V^T [B,C,N]
                                            half_t* __restrict__ Hb) {
    __shared__ half_t sB[2][64 * 256];
    const int NT = CC / 64;                         // 8
    int nt  = blockIdx.x % NT;
    int mtg = (blockIdx.x / NT) & 7;
    int b   = blockIdx.x / (NT * 8);
    int wv  = threadIdx.x >> 5;
    int m0  = (mtg * 8 + wv) * 16;
    int c0  = nt * 64;

    const half_t* A  = P  + (size_t)b * NN * NN + (size_t)m0 * NN;
    const half_t* Vt = vb + (size_t)b * CC * NN + (size_t)c0 * NN;

    async_issue_panel(sB[0], Vt, NN);
    v8f acc[4] = {};
    for (int h = 0; h < 4; ++h) {                   // K = 1024 in 4 chunks of 256
        async_wait();
        __syncthreads();
        if (h + 1 < 4) async_issue_panel(sB[(h + 1) & 1], Vt + (h + 1) * 256, NN);
        const half_t* Ah = A + h * 256;
        const half_t* Bp = sB[h & 1];
#pragma unroll 4
        for (int kk = 0; kk < 256; kk += 32) {
            v16h a = load_a_f16(Ah + kk, NN);
#pragma unroll
            for (int t = 0; t < 4; ++t) {
                v16h bf = load_b_t(Bp + (size_t)(t * 16) * 256 + kk, 256);
                acc[t] = wmma_f16(a, bf, acc[t]);
            }
        }
    }
    int lane = threadIdx.x & 31;
    int n = lane & 15, mb = (lane >> 4) * 8;
    half_t* Ho = Hb + (size_t)b * NN * CC;
#pragma unroll
    for (int t = 0; t < 4; ++t) {
#pragma unroll
        for (int rr = 0; rr < 8; ++rr) {
            Ho[(size_t)(m0 + mb + rr) * CC + (c0 + t * 16 + n)] = (half_t)acc[t][rr];
        }
    }
}

// =================== Kernel 6: proj GEMM + bias + residual -> out f32 [B,C,H,W] ===================
__global__ void __launch_bounds__(256) k_proj(const half_t* __restrict__ Hb,
                                              const half_t* __restrict__ wp,
                                              const float* __restrict__ proj_b,
                                              const float* __restrict__ x,
                                              float* __restrict__ out) {
    const int MT = NN / 16;   // 64
    const int NT = CC / 64;   // 8
    int wid = blockIdx.x * 8 + (threadIdx.x >> 5);
    int b  = wid / (MT * NT);
    int r  = wid % (MT * NT);
    int mt = r / NT, nt = r % NT;
    int m0 = mt * 16, n0 = nt * 64;

    const half_t* A = Hb + (size_t)b * NN * CC + (size_t)m0 * CC;
    v8f acc[4] = {};
    for (int c = 0; c < CC; c += 32) {
        v16h a = load_a_f16(A + c, CC);
#pragma unroll
        for (int t = 0; t < 4; ++t) {
            v16h bf = load_b_t(wp + (size_t)(n0 + t * 16) * CC + c, CC);
            acc[t] = wmma_f16(a, bf, acc[t]);
        }
    }
    int lane = threadIdx.x & 31;
    int n = lane & 15, mb = (lane >> 4) * 8;
#pragma unroll
    for (int t = 0; t < 4; ++t) {
        int cout = n0 + t * 16 + n;
        float bias = proj_b[cout];
#pragma unroll
        for (int rr = 0; rr < 8; ++rr) {
            int p = m0 + mb + rr;
            size_t idx = (size_t)b * CC * NN + (size_t)cout * NN + p;
            out[idx] = x[idx] + acc[t][rr] + bias;
        }
    }
}

extern "C" void kernel_launch(void* const* d_in, const int* in_sizes, int n_in,
                              void* d_out, int out_size, void* d_ws, size_t ws_size,
                              hipStream_t stream) {
    const float* x        = (const float*)d_in[0];
    const float* gn_gamma = (const float*)d_in[1];
    const float* gn_beta  = (const float*)d_in[2];
    const float* qkv_w    = (const float*)d_in[3];
    const float* qkv_b    = (const float*)d_in[4];
    const float* proj_w   = (const float*)d_in[5];
    const float* proj_b   = (const float*)d_in[6];
    float* out = (float*)d_out;

    // Workspace carve-up
    char* w = (char*)d_ws;
    half_t* wq = (half_t*)w; w += (size_t)OO * CC * 2;        // 1.5 MB
    half_t* wp = (half_t*)w; w += (size_t)CC * CC * 2;        // 0.5 MB
    half_t* hn = (half_t*)w; w += (size_t)BB * NN * CC * 2;   // 32 MB
    half_t* qb = (half_t*)w; w += (size_t)BB * NN * CC * 2;   // 32 MB
    half_t* kb = (half_t*)w; w += (size_t)BB * NN * CC * 2;   // 32 MB
    half_t* vb = (half_t*)w; w += (size_t)BB * NN * CC * 2;   // 32 MB  (V^T)
    float*  S  = (float*)w;  w += (size_t)BB * NN * NN * 4;   // 128 MB
    half_t* P  = (half_t*)w; w += (size_t)BB * NN * NN * 2;   // 64 MB
    half_t* Hb = (half_t*)w; w += (size_t)BB * NN * CC * 2;   // 32 MB

    k_cvt_weights<<<(OO * CC + 255) / 256, 256, 0, stream>>>(qkv_w, proj_w, wq, wp);
    k_groupnorm<<<BB * GG, 256, 0, stream>>>(x, gn_gamma, gn_beta, hn);
    k_qkv<<<BB * 8 * (OO / 64), 256, 0, stream>>>(hn, wq, qkv_b, qb, kb, vb);
    k_scores<<<BB * 8 * (NN / 64), 256, 0, stream>>>(qb, kb, S);
    k_softmax<<<BB * NN, 256, 0, stream>>>(S, P);
    k_pv<<<BB * 8 * (CC / 64), 256, 0, stream>>>(P, vb, Hb);
    k_proj<<<(BB * (NN / 16) * (CC / 64)) / 8, 256, 0, stream>>>(Hb, wp, proj_b, x, out);
}
